// GraphSENNPool_54589034332715
// MI455X (gfx1250) — matrix operand order repaired and checked by reading, not compiled, not run on hardware
//
#include <hip/hip_runtime.h>

// ---------------------------------------------------------------------------
// GraphSENNPool fused implementation for gfx1250 (MI455X), bf16 WMMA path.
// ---------------------------------------------------------------------------

typedef __bf16 bf16_t;
typedef __attribute__((ext_vector_type(16))) __bf16 v16bf;
typedef __attribute__((ext_vector_type(8)))  __bf16 v8bf;
typedef __attribute__((ext_vector_type(8)))  float  v8f;

#define DD  256      // input feature dim
#define CC  10       // classes
#define GG  512      // graphs
#define MB  32       // rows (nodes) per block
#define LDA 264      // LDS row stride in bf16 elements (256 + 8 pad)

// ---- fragment helpers (layouts per CDNA5 ISA 7.12.2) ----------------------

__device__ __forceinline__ v16bf mk16(v8bf lo, v8bf hi) {
  return __builtin_shufflevector(lo, hi, 0,1,2,3,4,5,6,7,8,9,10,11,12,13,14,15);
}

// A-matrix 16x32 bf16 fragment from LDS tile (row-major, stride LDA).
// lane L (mr=L&15, hi=L>>4) holds row M=mr with K = k0+hi*8+[0..8) (elems 0..7)
// and K = k0+16+hi*8+[0..8) (elems 8..15).
__device__ __forceinline__ v16bf load_a(const bf16_t* A, int m0, int k0, int lane) {
  const int mr = lane & 15, hi = lane >> 4;
  const bf16_t* p = A + (m0 + mr) * LDA + k0 + hi * 8;
  return mk16(*(const v8bf*)p, *(const v8bf*)(p + 16));
}

// B-matrix 32x16 bf16 fragment from transposed weights Bt[n][k] (k contiguous,
// row stride ldk). lane L (nc=L&15, hi=L>>4) holds col N=n0+nc with
// K = k0+hi*8+[0..8) (elems 0..7) and K = k0+16+hi*8+[0..8) (elems 8..15).
__device__ __forceinline__ v16bf load_b(const bf16_t* Bt, int ldk, int n0, int k0, int lane) {
  const int nc = lane & 15, hi = lane >> 4;
  const bf16_t* p = Bt + (size_t)(n0 + nc) * ldk + k0 + hi * 8;
  return mk16(*(const v8bf*)p, *(const v8bf*)(p + 16));
}

__device__ __forceinline__ v8f wmma_bf16(v16bf a, v16bf b, v8f c) {
  return __builtin_amdgcn_wmma_f32_16x16x32_bf16(false, a, false, b, (short)0, c,
                                                 false, false);
}

// 256-wide layer: D[M,256] = act(A[M,256] @ Bt^T + bias), D/A bf16 in LDS.
__device__ __forceinline__ void layer256(const bf16_t* A, const bf16_t* Bt,
                                         const float* __restrict__ bias,
                                         bf16_t* D, int w, int lane, bool relu) {
  for (int t = w; t < 32; t += 8) {                 // 2 M-tiles x 16 N-tiles
    const int m0 = (t >> 4) * 16, n0 = (t & 15) * 16;
    v8f c = {};
    for (int k0 = 0; k0 < 256; k0 += 32) {
      v16bf a = load_a(A, m0, k0, lane);
      v16bf b = load_b(Bt, 256, n0, k0, lane);
      c = wmma_bf16(a, b, c);
    }
    const int nc = lane & 15, hi = lane >> 4;
    const float bv = bias[n0 + nc];
    #pragma unroll
    for (int v = 0; v < 8; v++) {
      float r = c[v] + bv;
      if (relu) r = fmaxf(r, 0.0f);
      D[(m0 + v + 8 * hi) * LDA + n0 + nc] = (bf16_t)r;
    }
  }
}

// Head layer: S[M,16] = A[M,256] @ Bt^T + bias (cols >= CC are zero-padded).
__device__ __forceinline__ void layerHead(const bf16_t* A, const bf16_t* Bt,
                                          const float* __restrict__ bias,
                                          float* S, int m0, int lane) {
  v8f c = {};
  for (int k0 = 0; k0 < 256; k0 += 32) {
    v16bf a = load_a(A, m0, k0, lane);
    v16bf b = load_b(Bt, 256, 0, k0, lane);
    c = wmma_bf16(a, b, c);
  }
  const int nc = lane & 15, hi = lane >> 4;
  const float bv = (nc < CC) ? bias[nc] : 0.0f;
  #pragma unroll
  for (int v = 0; v < 8; v++)
    S[(m0 + v + 8 * hi) * 16 + nc] = c[v] + bv;
}

// ---- pre-pass kernels ------------------------------------------------------

__global__ void zero2_kernel(float* __restrict__ a, int na,
                             float* __restrict__ b, int nb) {
  const int i = blockIdx.x * blockDim.x + threadIdx.x;
  if (i < na) a[i] = 0.0f;
  if (i < nb) b[i] = 0.0f;
}

// Wt[n][k] = bf16(W[k][n]); n >= Nsrc zero-padded. W is [K][Nsrc].
__global__ void convert_wt_kernel(const float* __restrict__ W,
                                  bf16_t* __restrict__ Wt,
                                  int K, int Nsrc, int Npad) {
  const int idx = blockIdx.x * blockDim.x + threadIdx.x;
  if (idx >= Npad * K) return;
  const int n = idx / K, k = idx - n * K;
  const float v = (n < Nsrc) ? W[(size_t)k * Nsrc + n] : 0.0f;
  Wt[idx] = (bf16_t)v;
}

// pooled[g, :] = segment_sum(x) over sorted batch. One thread per feature,
// one block per 256-row chunk; atomic flush only at segment boundaries.
__global__ __launch_bounds__(256)
void pool_seg_kernel(const float* __restrict__ x, const int* __restrict__ batch,
                     float* __restrict__ pooled, int N) {
  const int f  = threadIdx.x;        // feature 0..255
  const int r0 = blockIdx.x * 256;
  float acc = 0.0f;
  int cur = -1;
  for (int i = 0; i < 256; i++) {
    const int r = r0 + i;
    if (r >= N) break;
    const int g = batch[r];
    if (g != cur) {
      if (cur >= 0) atomicAdd(&pooled[(size_t)cur * DD + f], acc);
      cur = g; acc = 0.0f;
    }
    acc += x[(size_t)r * DD + f];
  }
  if (cur >= 0) atomicAdd(&pooled[(size_t)cur * DD + f], acc);
}

// ---- fused main kernel -----------------------------------------------------

__global__ __launch_bounds__(256)
void senn_fused_kernel(const float* __restrict__ x, const int* __restrict__ batch,
                       const float* __restrict__ pooled,
                       const bf16_t* __restrict__ W0t, const float* __restrict__ b0,
                       const bf16_t* __restrict__ W1t, const float* __restrict__ b1,
                       const bf16_t* __restrict__ W2t, const float* __restrict__ b2,
                       const bf16_t* __restrict__ tW0t, const float* __restrict__ tb0,
                       const bf16_t* __restrict__ tW1t, const float* __restrict__ tb1,
                       float* __restrict__ outG, float* __restrict__ thetaOut,
                       float* __restrict__ hOut, int N) {
  __shared__ bf16_t xA[MB * LDA];    // x tile (bf16)
  __shared__ bf16_t pA[MB * LDA];    // pooled[batch] tile, later reused as h1
  __shared__ bf16_t bufA[MB * LDA];  // theta hidden t1
  __shared__ float  hS[MB * 16];
  __shared__ float  tS[MB * 16];

  const int tid  = threadIdx.x;
  const int w    = tid >> 5;
  const int lane = tid & 31;
  const int r0   = blockIdx.x * MB;

  // ---- stage x and pooled[batch] tiles as bf16 ----
  for (int idx = tid; idx < MB * DD; idx += 256) {
    const int r = idx >> 8, c = idx & 255;
    const int row = r0 + r;
    float xv = 0.0f, pv = 0.0f;
    if (row < N) {
      xv = x[(size_t)row * DD + c];
      pv = pooled[(size_t)batch[row] * DD + c];
    }
    xA[r * LDA + c] = (bf16_t)xv;
    pA[r * LDA + c] = (bf16_t)pv;
  }
  __syncthreads();

  // ---- theta layer 0: t1 = relu([x | pooled] @ tW0 + tb0), K = 512 ----
  for (int t = w; t < 32; t += 8) {
    const int m0 = (t >> 4) * 16, n0 = (t & 15) * 16;
    v8f c = {};
    for (int k0 = 0; k0 < 256; k0 += 32) {
      v16bf a = load_a(xA, m0, k0, lane);
      v16bf b = load_b(tW0t, 512, n0, k0, lane);
      c = wmma_bf16(a, b, c);
    }
    for (int k0 = 0; k0 < 256; k0 += 32) {
      v16bf a = load_a(pA, m0, k0, lane);
      v16bf b = load_b(tW0t, 512, n0, 256 + k0, lane);
      c = wmma_bf16(a, b, c);
    }
    const int nc = lane & 15, hi = lane >> 4;
    const float bv = tb0[n0 + nc];
    #pragma unroll
    for (int v = 0; v < 8; v++) {
      float r = fmaxf(c[v] + bv, 0.0f);
      bufA[(m0 + v + 8 * hi) * LDA + n0 + nc] = (bf16_t)r;
    }
  }
  __syncthreads();

  // ---- h layer 0: h1 = relu(x @ W0 + b0) -> pA (pooled tile now dead) ----
  layer256(xA, W0t, b0, pA, w, lane, true);
  __syncthreads();

  // ---- h layer 1: h2 = relu(h1 @ W1 + b1) -> xA (x tile now dead) ----
  layer256(pA, W1t, b1, xA, w, lane, true);
  __syncthreads();

  // ---- heads: h = h2 @ W2 + b2 ; theta = t1 @ tW1 + tb1 ----
  if      (w == 0) layerHead(xA,   W2t,  b2,  hS,  0, lane);
  else if (w == 1) layerHead(xA,   W2t,  b2,  hS, 16, lane);
  else if (w == 2) layerHead(bufA, tW1t, tb1, tS,  0, lane);
  else if (w == 3) layerHead(bufA, tW1t, tb1, tS, 16, lane);
  __syncthreads();

  // ---- epilogue: store h, theta; segment-sum h*theta into out[G,C] ----
  for (int idx = tid; idx < MB * CC; idx += 256) {
    const int r = idx / CC, c = idx - r * CC;
    const int row = r0 + r;
    if (row < N) {
      const float hv = hS[r * 16 + c];
      const float tv = tS[r * 16 + c];
      hOut[(size_t)row * CC + c]     = hv;
      thetaOut[(size_t)row * CC + c] = tv;
      atomicAdd(&outG[(size_t)batch[row] * CC + c], hv * tv);
    }
  }
}

// ---------------------------------------------------------------------------

extern "C" void kernel_launch(void* const* d_in, const int* in_sizes, int n_in,
                              void* d_out, int out_size, void* d_ws, size_t ws_size,
                              hipStream_t stream) {
  (void)n_in; (void)out_size; (void)ws_size;

  const float* x    = (const float*)d_in[0];
  const int*   batch= (const int*)  d_in[1];
  const float* hW0  = (const float*)d_in[2];
  const float* hb0  = (const float*)d_in[3];
  const float* hW1  = (const float*)d_in[4];
  const float* hb1  = (const float*)d_in[5];
  const float* hW2  = (const float*)d_in[6];
  const float* hb2  = (const float*)d_in[7];
  const float* tW0  = (const float*)d_in[8];
  const float* tb0  = (const float*)d_in[9];
  const float* tW1  = (const float*)d_in[10];
  const float* tb1  = (const float*)d_in[11];

  const int N = in_sizes[0] / DD;   // 300000

  // workspace layout (256B aligned offsets)
  char*   ws     = (char*)d_ws;
  float*  pooled = (float*) (ws + 0);          // 512*256*4   = 524288 B
  bf16_t* W0t    = (bf16_t*)(ws + 524288);     // 256x256 bf16 = 131072 B
  bf16_t* W1t    = (bf16_t*)(ws + 655360);     // 131072 B
  bf16_t* W2t    = (bf16_t*)(ws + 786432);     // 16x256 bf16  = 8192 B
  bf16_t* tW0t   = (bf16_t*)(ws + 794624);     // 256x512 bf16 = 262144 B
  bf16_t* tW1t   = (bf16_t*)(ws + 1056768);    // 8192 B       (end ~1.02 MB)

  // output layout: out[512*10] | theta[N*10] | h[N*10]
  float* outG  = (float*)d_out;
  float* thOut = outG + (size_t)GG * CC;
  float* hOut  = thOut + (size_t)N * CC;

  // 1) zero accumulators (out + pooled)
  zero2_kernel<<<(GG * DD + 255) / 256, 256, 0, stream>>>(outG, GG * CC,
                                                          pooled, GG * DD);
  // 2) convert + transpose weights to bf16
  convert_wt_kernel<<<(256 * 256 + 255) / 256, 256, 0, stream>>>(hW0, W0t, 256, 256, 256);
  convert_wt_kernel<<<(256 * 256 + 255) / 256, 256, 0, stream>>>(hW1, W1t, 256, 256, 256);
  convert_wt_kernel<<<(16  * 256 + 255) / 256, 256, 0, stream>>>(hW2, W2t, 256, CC, 16);
  convert_wt_kernel<<<(256 * 512 + 255) / 256, 256, 0, stream>>>(tW0, tW0t, 512, 256, 256);
  convert_wt_kernel<<<(16  * 256 + 255) / 256, 256, 0, stream>>>(tW1, tW1t, 256, CC, 16);
  // 3) per-graph pooling of x
  pool_seg_kernel<<<(N + 255) / 256, 256, 0, stream>>>(x, batch, pooled, N);
  // 4) fused WMMA pipeline
  senn_fused_kernel<<<(N + MB - 1) / MB, 256, 0, stream>>>(
      x, batch, pooled, W0t, hb0, W1t, hb1, W2t, hb2,
      tW0t, tb0, tW1t, tb1, outG, thOut, hOut, N);
}
